// ScaledDotProductAttention_43739946942936
// MI455X (gfx1250) — matrix operand order repaired and checked by reading, not compiled, not run on hardware
//
#include <hip/hip_runtime.h>

// ---------------------------------------------------------------------------
// Flash-attention forward for MI455X (gfx1250, wave32, WMMA bf16).
// B=8, NQ=2048, SK=2048, D=128, fp32 in/out.
// Block = 128 threads (4 waves); wave w owns q rows [q0+16w, q0+16w+16).
// Streams keys in 32-wide blocks through LDS; scores never touch HBM.
// fp32->bf16 via native casts so the backend can pick v_cvt_pk_bf16_f32.
// ---------------------------------------------------------------------------

typedef __attribute__((ext_vector_type(16))) __bf16        v16bf;
typedef __attribute__((ext_vector_type(4)))  __bf16        v4bf;
typedef __attribute__((ext_vector_type(8)))  float         v8f;
typedef __attribute__((ext_vector_type(8)))  unsigned int  v8u;

static constexpr int Bc  = 8;
static constexpr int NQc = 2048;
static constexpr int SKc = 2048;
static constexpr int Dc  = 128;

static __device__ __forceinline__ __bf16 f2bf(float f) {
  return (__bf16)f;  // RNE; backend selects native bf16 cvt on gfx1250
}
static __device__ __forceinline__ unsigned short f2bfu(float f) {
  __bf16 b = (__bf16)f;
  return __builtin_bit_cast(unsigned short, b);
}

struct U32B { uint4 lo, hi; };  // 32-byte carrier for bit_cast to v16bf

__global__ __launch_bounds__(128) void sdpa_flash_wmma(
    const float* __restrict__ Q, const float* __restrict__ K,
    const float* __restrict__ V, const unsigned char* __restrict__ Msk,
    const float* __restrict__ W, float* __restrict__ Out) {
  // LDS staging: K tile row-major bf16, V tile transposed bf16, P per-wave.
  __shared__ unsigned short sK [32 * 128];       // [kk][d]    8 KB
  __shared__ unsigned short sVt[128 * 32];       // [d][kk]    8 KB
  __shared__ unsigned short sP [4][16 * 32];     // [m][kk]    4 KB

  const int tid  = threadIdx.x;
  const int lane = tid & 31;
  const int wv   = tid >> 5;        // wave id 0..3
  const int h    = lane >> 4;       // half-wave select
  const int nn   = lane & 15;       // N / column index

  const int b   = blockIdx.x >> 5;          // NQ/64 = 32 q-blocks per batch
  const int q0  = (blockIdx.x & 31) * 64;
  const int q0w = q0 + wv * 16;             // this wave's first q row

  const float scale = 0.08838834764831845f; // 1/sqrt(128), folded into Q

  // ---- Load this wave's 16x128 Q tile into A-operand layout (bf16) ----
  // A 16x32 layout: lane m = lane&15; elems i -> K = 16h+i
  //                 (per D-chunk of 32).
  v16bf aq[4];
  {
    const int row = q0w + nn;
    const float* qrow = Q + ((size_t)b * NQc + row) * Dc;
#pragma unroll
    for (int c = 0; c < 4; ++c) {
      const float* p0 = qrow + 32 * c + 16 * h;
      const float* p1 = qrow + 32 * c + 16 * h + 8;
      float4 x0 = ((const float4*)p0)[0];
      float4 x1 = ((const float4*)p0)[1];
      float4 x2 = ((const float4*)p1)[0];
      float4 x3 = ((const float4*)p1)[1];
      v16bf a;
      a[0]  = f2bf(x0.x * scale); a[1]  = f2bf(x0.y * scale);
      a[2]  = f2bf(x0.z * scale); a[3]  = f2bf(x0.w * scale);
      a[4]  = f2bf(x1.x * scale); a[5]  = f2bf(x1.y * scale);
      a[6]  = f2bf(x1.z * scale); a[7]  = f2bf(x1.w * scale);
      a[8]  = f2bf(x2.x * scale); a[9]  = f2bf(x2.y * scale);
      a[10] = f2bf(x2.z * scale); a[11] = f2bf(x2.w * scale);
      a[12] = f2bf(x3.x * scale); a[13] = f2bf(x3.y * scale);
      a[14] = f2bf(x3.z * scale); a[15] = f2bf(x3.w * scale);
      aq[c] = a;
    }
  }

  // Flash state: O (16x128 fp32 as 8 C/D tiles), running max / sum per row.
  v8f   O[8] = {};
  float Mx[8], L[8];
#pragma unroll
  for (int v = 0; v < 8; ++v) { Mx[v] = -1e30f; L[v] = 0.0f; }

  for (int k0 = 0; k0 < SKc; k0 += 32) {
    // ---- Cooperative load: K -> sK (row major), V -> sVt (transposed) ----
    {
      const float* Kg = K + ((size_t)b * SKc + k0) * Dc;
      const float* Vg = V + ((size_t)b * SKc + k0) * Dc;
#pragma unroll
      for (int r = 0; r < 8; ++r) {
        int idx = tid + 128 * r;      // float4 index 0..1023
        int e   = idx * 4;            // element index 0..4095
        int kk  = e >> 7;
        int d   = e & 127;
        float4 kx = ((const float4*)Kg)[idx];
        v4bf pk;
        pk[0] = f2bf(kx.x); pk[1] = f2bf(kx.y);
        pk[2] = f2bf(kx.z); pk[3] = f2bf(kx.w);
        *(v4bf*)&sK[e] = pk;          // 8-byte aligned LDS store
        float4 vx = ((const float4*)Vg)[idx];
        sVt[(d + 0) * 32 + kk] = f2bfu(vx.x);
        sVt[(d + 1) * 32 + kk] = f2bfu(vx.y);
        sVt[(d + 2) * 32 + kk] = f2bfu(vx.z);
        sVt[(d + 3) * 32 + kk] = f2bfu(vx.w);
      }
    }
    __syncthreads();

    // ---- S = Q * K^T : two 16-key N-tiles, 4 D-chunks each ----
    // B 32x16 layout: lane col n = lane&15; elems i -> Kdim = 16h + i
    // (contiguous 16 bf16 = one 32B LDS read). Pipeline loads 1 chunk ahead.
    v8f S0 = {}, S1 = {};
    {
      v16bf b0 = __builtin_bit_cast(
          v16bf, *(const v8u*)&sK[(0  + nn) * 128 + 16 * h]);
      v16bf b1 = __builtin_bit_cast(
          v16bf, *(const v8u*)&sK[(16 + nn) * 128 + 16 * h]);
#pragma unroll
      for (int c = 0; c < 4; ++c) {
        v16bf nb0, nb1;
        if (c < 3) {
          nb0 = __builtin_bit_cast(
              v16bf, *(const v8u*)&sK[(0  + nn) * 128 + 32 * (c + 1) + 16 * h]);
          nb1 = __builtin_bit_cast(
              v16bf, *(const v8u*)&sK[(16 + nn) * 128 + 32 * (c + 1) + 16 * h]);
        }
        S0 = __builtin_amdgcn_wmma_f32_16x16x32_bf16(
            false, aq[c], false, b0, (short)0, S0, false, false);
        S1 = __builtin_amdgcn_wmma_f32_16x16x32_bf16(
            false, aq[c], false, b1, (short)0, S1, false, false);
        b0 = nb0; b1 = nb1;
      }
    }

    // ---- weights, mask, online softmax (C/D layout: m=8h+v, n=nn) ----
    const size_t wbase =
        ((size_t)b * NQc + (q0w + 8 * h)) * (size_t)SKc + k0 + nn;
    if (k0 + 32 < SKc) {  // prefetch next block's weight stream
      __builtin_prefetch(&W[wbase + 32], 0, 0);
      __builtin_prefetch(&Msk[wbase + 32], 0, 0);
    }
    float e0[8], e1[8];
#pragma unroll
    for (int v = 0; v < 8; ++v) {
      size_t off = wbase + (size_t)v * SKc;
      float w0 = W[off];          unsigned char m0 = Msk[off];
      float w1 = W[off + 16];     unsigned char m1 = Msk[off + 16];
      float s0 = S0[v] * w0;      if (m0) s0 = -1e30f;   // scale already in Q
      float s1 = S1[v] * w1;      if (m1) s1 = -1e30f;
      // row max across the 16 lanes of this half-wave
      float lmax = fmaxf(s0, s1);
      lmax = fmaxf(lmax, __shfl_xor(lmax, 1, 16));
      lmax = fmaxf(lmax, __shfl_xor(lmax, 2, 16));
      lmax = fmaxf(lmax, __shfl_xor(lmax, 4, 16));
      lmax = fmaxf(lmax, __shfl_xor(lmax, 8, 16));
      float mnew = fmaxf(Mx[v], lmax);
      float rsc  = __expf(Mx[v] - mnew);
      Mx[v] = mnew;
      float ee0 = __expf(s0 - mnew);
      float ee1 = __expf(s1 - mnew);
      float ls  = ee0 + ee1;
      ls += __shfl_xor(ls, 1, 16);
      ls += __shfl_xor(ls, 2, 16);
      ls += __shfl_xor(ls, 4, 16);
      ls += __shfl_xor(ls, 8, 16);
      L[v] = L[v] * rsc + ls;
      e0[v] = ee0; e1[v] = ee1;
#pragma unroll
      for (int j = 0; j < 8; ++j) O[j][v] *= rsc;   // rescale accumulator
    }

    // ---- stage P (bf16, row major 16x32) for C/D -> A layout conversion ----
#pragma unroll
    for (int v = 0; v < 8; ++v) {
      int m = 8 * h + v;
      sP[wv][m * 32 + nn]      = f2bfu(e0[v]);
      sP[wv][m * 32 + 16 + nn] = f2bfu(e1[v]);
    }
    __syncthreads();

    // ---- O += P * V : read P in A layout, V^T tiles in B layout ----
    v16bf ap;
    {
      uint4 lo = *(const uint4*)&sP[wv][nn * 32 + 16 * h];       // kk = 16h+0..7
      uint4 hi = *(const uint4*)&sP[wv][nn * 32 + 16 * h + 8];   // kk = 16h+8..15
      U32B u; u.lo = lo; u.hi = hi;
      ap = __builtin_bit_cast(v16bf, u);
    }
    {
      v16bf bv = __builtin_bit_cast(
          v16bf, *(const v8u*)&sVt[nn * 32 + 16 * h]);
#pragma unroll
      for (int j = 0; j < 8; ++j) {
        v16bf nbv;
        if (j < 7) {
          nbv = __builtin_bit_cast(
              v16bf, *(const v8u*)&sVt[(16 * (j + 1) + nn) * 32 + 16 * h]);
        }
        O[j] = __builtin_amdgcn_wmma_f32_16x16x32_bf16(
            false, ap, false, bv, (short)0, O[j], false, false);
        bv = nbv;
      }
    }
    __syncthreads();   // protect sK/sVt/sP for next iteration
  }

  // ---- normalize and store (coalesced across nn) ----
#pragma unroll
  for (int v = 0; v < 8; ++v) {
    float inv = 1.0f / L[v];
    int row = q0w + 8 * h + v;
    float* orow = Out + ((size_t)b * NQc + row) * Dc;
#pragma unroll
    for (int j = 0; j < 8; ++j) orow[16 * j + nn] = O[j][v] * inv;
  }
}

extern "C" void kernel_launch(void* const* d_in, const int* in_sizes, int n_in,
                              void* d_out, int out_size, void* d_ws,
                              size_t ws_size, hipStream_t stream) {
  const float*         Q   = (const float*)d_in[0];
  const float*         K   = (const float*)d_in[1];
  const float*         V   = (const float*)d_in[2];
  const unsigned char* Msk = (const unsigned char*)d_in[3];  // jnp.bool_ -> 1 byte
  const float*         W   = (const float*)d_in[4];
  float*               Out = (float*)d_out;

  dim3 grid(Bc * (NQc / 64));  // 256 blocks; 4 waves (128 thr) each
  sdpa_flash_wmma<<<grid, 128, 0, stream>>>(Q, K, V, Msk, W, Out);

  (void)in_sizes; (void)n_in; (void)out_size; (void)d_ws; (void)ws_size;
}
